// Decoder_20495583936950
// MI455X (gfx1250) — matrix operand order repaired and checked by reading, not compiled, not run on hardware
//
#include <hip/hip_runtime.h>
#include <hip/hip_bf16.h>

// ---------------------------------------------------------------------------
// Types / helpers
// ---------------------------------------------------------------------------
typedef unsigned short u16;
typedef __attribute__((ext_vector_type(16))) __bf16 v16bf;
typedef __attribute__((ext_vector_type(8)))  float  v8f;

union Frag16 {           // 16 bf16 = one WMMA A/B fragment (8 VGPRs)
    v16bf v;
    uint4 q[2];
};

__device__ __forceinline__ u16 f2bf(float f) {
    unsigned int u = __builtin_bit_cast(unsigned int, f);
    unsigned int r = u + 0x7FFFu + ((u >> 16) & 1u);   // round-to-nearest-even
    return (u16)(r >> 16);
}

__device__ __forceinline__ float gelu_f(float x) {
    float x3 = x * x * x;
    return 0.5f * x * (1.f + tanhf(0.7978845608f * (x + 0.044715f * x3)));
}

__device__ __forceinline__ v8f wmma_bf16(v16bf a, v16bf b, v8f c) {
    // D = A(16x32 bf16) * B(32x16 bf16) + C(16x16 f32)
    return __builtin_amdgcn_wmma_f32_16x16x32_bf16(
        /*neg_a=*/false, a, /*neg_b=*/false, b,
        /*c_mod=*/(short)0, c, /*reuse_a=*/false, /*reuse_b=*/false);
}

// async global->LDS copy of 16 bytes (CDNA5 ASYNCcnt path); lds offset is the
// low 32 bits of the generic shared-space address (aperture {SHARED_BASE, off}).
__device__ __forceinline__ void async_copy_b128(void* lds_dst, const void* gsrc) {
    unsigned lds_off = (unsigned)(unsigned long long)lds_dst;
    asm volatile("global_load_async_to_lds_b128 %0, %1, off"
                 :: "v"(lds_off), "v"(gsrc) : "memory");
}
__device__ __forceinline__ void wait_asynccnt0() {
    asm volatile("s_wait_asynccnt 0" ::: "memory");
}

// ---------------------------------------------------------------------------
// Problem constants
// ---------------------------------------------------------------------------
#define D_MODEL 256
#define D_PATCH 768
#define K_CODES 8192
#define SEQLEN  1024
#define N_HEADS 8
#define DHEAD   32
#define N_BLOCKS 4
#define DFF     1024
#define BATCH   16
#define MTOK    (BATCH * SEQLEN)   // 16384 rows

// ---------------------------------------------------------------------------
// fp32 -> bf16 convert
// ---------------------------------------------------------------------------
__global__ void cvt_f32_bf16_kernel(const float* __restrict__ in,
                                    u16* __restrict__ out) {
    int i = blockIdx.x * 256 + threadIdx.x;
    out[i] = f2bf(in[i]);
}

// ---------------------------------------------------------------------------
// codebook row norms ||c_k||^2  (one wave per row, 8 rows per block)
// ---------------------------------------------------------------------------
__global__ void cb_norms_kernel(const float* __restrict__ cb,
                                float* __restrict__ norms) {
    int lane = threadIdx.x & 31;
    int wave = threadIdx.x >> 5;
    int row  = blockIdx.x * 8 + wave;
    const float* r = cb + (size_t)row * D_MODEL;
    float s = 0.f;
#pragma unroll
    for (int i = 0; i < 8; ++i) {
        float v = r[lane + i * 32];
        s += v * v;
    }
#pragma unroll
    for (int off = 16; off > 0; off >>= 1) s += __shfl_xor(s, off);
    if (lane == 0) norms[row] = s;
}

// ---------------------------------------------------------------------------
// VQ argmin: per token, argmin_k (||c_k||^2 - 2 * ze.c_k) via bf16 WMMA
// block = 128 threads (4 waves), 16 token rows per block; waves stride N tiles
// ---------------------------------------------------------------------------
__global__ void vq_argmin_kernel(const u16* __restrict__ zeb,
                                 const u16* __restrict__ cbb,
                                 const float* __restrict__ norms,
                                 int* __restrict__ idxOut) {
    __shared__ float sv[4][16];
    __shared__ int   si[4][16];

    int lane = threadIdx.x & 31;
    int wave = threadIdx.x >> 5;
    int rowBase = blockIdx.x * 16;

    // preload A fragments (ze rows, 8 K-steps of 32) into registers
    Frag16 a[8];
    {
        int m  = rowBase + (lane & 15);
        int kc = (lane >> 4) * 8;
        const u16* arow = zeb + (size_t)m * D_MODEL;
#pragma unroll
        for (int ks = 0; ks < 8; ++ks) {
            a[ks].q[0] = *(const uint4*)(arow + ks * 32 + kc);
            a[ks].q[1] = *(const uint4*)(arow + ks * 32 + kc + 16);
        }
    }

    float bestV[8];
    int   bestI[8];
#pragma unroll
    for (int v = 0; v < 8; ++v) { bestV[v] = 3.4e38f; bestI[v] = 0; }

    // 8192 codes / 16-wide N tiles = 512 tiles; 4 waves stride them
    for (int t = wave; t < K_CODES / 16; t += 4) {
        int n0   = t * 16;
        int code = n0 + (lane & 15);
        const u16* brow = cbb + (size_t)code * D_MODEL + (lane >> 4) * 8;

        // overlap: pull next tile of this wave toward L2/WGP while we compute
        __builtin_prefetch(brow + 4 * 16 * D_MODEL, 0, 1);

        v8f acc;
#pragma unroll
        for (int v = 0; v < 8; ++v) acc[v] = 0.f;

        Frag16 bf;
#pragma unroll
        for (int ks = 0; ks < 8; ++ks) {
            bf.q[0] = *(const uint4*)(brow + ks * 32);
            bf.q[1] = *(const uint4*)(brow + ks * 32 + 16);
            acc = wmma_bf16(a[ks].v, bf.v, acc);
        }

        float nrm = norms[code];
#pragma unroll
        for (int v = 0; v < 8; ++v) {
            float d = nrm - 2.f * acc[v];
            if (d < bestV[v]) { bestV[v] = d; bestI[v] = code; }
        }
    }

    // reduce (min, argmin) across the 16 lanes of each half
#pragma unroll
    for (int off = 8; off > 0; off >>= 1) {
#pragma unroll
        for (int v = 0; v < 8; ++v) {
            float ov = __shfl_xor(bestV[v], off);
            int   oi = __shfl_xor(bestI[v], off);
            if (ov < bestV[v]) { bestV[v] = ov; bestI[v] = oi; }
        }
    }
    if ((lane & 15) == 0) {
        int half = (lane >= 16) ? 8 : 0;
#pragma unroll
        for (int v = 0; v < 8; ++v) {
            sv[wave][v + half] = bestV[v];
            si[wave][v + half] = bestI[v];
        }
    }
    __syncthreads();
    if (threadIdx.x < 16) {
        float bv = sv[0][threadIdx.x];
        int   bi = si[0][threadIdx.x];
#pragma unroll
        for (int w = 1; w < 4; ++w)
            if (sv[w][threadIdx.x] < bv) { bv = sv[w][threadIdx.x]; bi = si[w][threadIdx.x]; }
        idxOut[rowBase + threadIdx.x] = bi;
    }
}

// ---------------------------------------------------------------------------
// x = codebook[idx] + pos_emb   (fp32)
// ---------------------------------------------------------------------------
__global__ void gather_pos_kernel(const int* __restrict__ idx,
                                  const float* __restrict__ cb,
                                  const float* __restrict__ pos,
                                  float* __restrict__ x) {
    int t   = blockIdx.x * 256 + threadIdx.x;     // over 16384*256
    int d   = t & (D_MODEL - 1);
    int tok = t >> 8;
    int s   = tok & (SEQLEN - 1);
    x[t] = cb[(size_t)idx[tok] * D_MODEL + d] + pos[(size_t)s * D_MODEL + d];
}

// ---------------------------------------------------------------------------
// LayerNorm over D=256, output bf16 (one wave per row, 8 rows/block)
// ---------------------------------------------------------------------------
__global__ void layernorm_kernel(const float* __restrict__ x,
                                 const float* __restrict__ sc,
                                 const float* __restrict__ bi,
                                 u16* __restrict__ out) {
    int lane = threadIdx.x & 31;
    int wave = threadIdx.x >> 5;
    int row  = blockIdx.x * 8 + wave;
    const float* xr = x + (size_t)row * D_MODEL;

    float v[8], s = 0.f, s2 = 0.f;
#pragma unroll
    for (int i = 0; i < 8; ++i) {
        v[i] = xr[lane + i * 32];
        s  += v[i];
        s2 += v[i] * v[i];
    }
#pragma unroll
    for (int off = 16; off > 0; off >>= 1) {
        s  += __shfl_xor(s,  off);
        s2 += __shfl_xor(s2, off);
    }
    float mean = s * (1.f / D_MODEL);
    float var  = s2 * (1.f / D_MODEL) - mean * mean;
    float inv  = rsqrtf(var + 1e-5f);
#pragma unroll
    for (int i = 0; i < 8; ++i) {
        int d = lane + i * 32;
        out[(size_t)row * D_MODEL + d] = f2bf((v[i] - mean) * inv * sc[d] + bi[d]);
    }
}

// ---------------------------------------------------------------------------
// Generic bf16 WMMA GEMM: C[M,N] = act(A[M,K] @ B[K,N] + bias [+ resid])
// block = 128 thr (4 waves, 2x2 wave grid), 64x64 block tile, K step 32.
// A tile staged via async global->LDS (ASYNCcnt path, no VGPR round-trip);
// B staged transposed (NxK) so both fragments are contiguous ds_load_b128
// per the CDNA5 WMMA VGPR layouts.
// ---------------------------------------------------------------------------
template <bool RESID, bool GELU_ACT, bool OUTB, bool OUTF>
__global__ void gemm_bf16_kernel(const u16* __restrict__ A,
                                 const u16* __restrict__ B,
                                 const float* __restrict__ bias,
                                 const float* __restrict__ resid,
                                 float* __restrict__ outF,
                                 u16* __restrict__ outB,
                                 int M, int N, int K) {
    __shared__ u16 As[64 * 32];    // [m][k]
    __shared__ u16 Bs[64 * 32];    // [n][k]  (transposed)

    int tid  = threadIdx.x;
    int lane = tid & 31;
    int wave = tid >> 5;
    int wr   = wave >> 1;          // wave row (0..1) -> 32 M rows
    int wc   = wave & 1;           // wave col (0..1) -> 32 N cols
    int mBlock = blockIdx.y * 64;
    int nBlock = blockIdx.x * 64;

    v8f acc[2][2];
#pragma unroll
    for (int i = 0; i < 2; ++i)
#pragma unroll
        for (int j = 0; j < 2; ++j)
#pragma unroll
            for (int v = 0; v < 8; ++v) acc[i][j][v] = 0.f;

    for (int k0 = 0; k0 < K; k0 += 32) {
        // stage A tile 64x32: two async b128 copies per thread (ASYNCcnt)
        {
            int e = tid * 16;
            int r = e >> 5, c = e & 31;
            const u16* src = A + (size_t)(mBlock + r) * K + k0 + c;
            u16* dst = As + r * 32 + c;
            async_copy_b128(dst, src);
            async_copy_b128(dst + 8, src + 8);
        }
        // stage B tile 32x64 transposed into Bs[n][k]
#pragma unroll
        for (int i = 0; i < 16; ++i) {
            int e  = tid + i * 128;
            int kk = e >> 6, nn = e & 63;
            Bs[nn * 32 + kk] = B[(size_t)(k0 + kk) * N + nBlock + nn];
        }
        wait_asynccnt0();
        __syncthreads();

        Frag16 af[2], bf[2];
        {
            int m0 = wr * 32 + (lane & 15);
            int kc = (lane >> 4) * 8;
#pragma unroll
            for (int i = 0; i < 2; ++i) {
                const u16* p = As + (m0 + i * 16) * 32 + kc;
                af[i].q[0] = *(const uint4*)p;
                af[i].q[1] = *(const uint4*)(p + 16);
            }
            int n0 = wc * 32 + (lane & 15);
            int kb = (lane >> 4) * 8;
#pragma unroll
            for (int j = 0; j < 2; ++j) {
                const u16* p = Bs + (n0 + j * 16) * 32 + kb;
                bf[j].q[0] = *(const uint4*)p;
                bf[j].q[1] = *(const uint4*)(p + 16);
            }
        }
#pragma unroll
        for (int i = 0; i < 2; ++i)
#pragma unroll
            for (int j = 0; j < 2; ++j)
                acc[i][j] = wmma_bf16(af[i].v, bf[j].v, acc[i][j]);
        __syncthreads();
    }

    // epilogue: C layout — lane<16: N=lane, M=vgpr; lane>=16: N=lane-16, M=8+vgpr
    int rowOff = (lane >= 16) ? 8 : 0;
#pragma unroll
    for (int i = 0; i < 2; ++i) {
#pragma unroll
        for (int j = 0; j < 2; ++j) {
            int ncol = nBlock + wc * 32 + j * 16 + (lane & 15);
            float bb = bias[ncol];
#pragma unroll
            for (int v = 0; v < 8; ++v) {
                int mrow = mBlock + wr * 32 + i * 16 + v + rowOff;
                float val = acc[i][j][v] + bb;
                if (RESID)    val += resid[(size_t)mrow * N + ncol];
                if (GELU_ACT) val  = gelu_f(val);
                if (OUTF) outF[(size_t)mrow * N + ncol] = val;
                if (OUTB) outB[(size_t)mrow * N + ncol] = f2bf(val);
            }
        }
    }
}

// ---------------------------------------------------------------------------
// Flash attention: one wave per (b, h, 16-query tile); Dh=32 == WMMA K.
// qkv bf16 layout [B*S, 768] (q|k|v concat); out bf16 [B*S, 256].
// ---------------------------------------------------------------------------
__global__ void attention_kernel(const u16* __restrict__ qkv,
                                 u16* __restrict__ o) {
    __shared__ u16 p_lds[16 * 32];   // P tile row-major [q][key]
    __shared__ u16 vt_lds[32 * 32];  // V tile transposed [dh][key]

    int lane = threadIdx.x;
    int qt = blockIdx.x, h = blockIdx.y, b = blockIdx.z;
    int s0 = qt * 16;
    const u16* base = qkv + (size_t)b * SEQLEN * (3 * D_MODEL);

    // Q A-fragment
    Frag16 qf;
    {
        int m  = s0 + (lane & 15);
        int dh = (lane >> 4) * 8;
        const u16* qp = base + (size_t)m * (3 * D_MODEL) + h * DHEAD + dh;
        qf.q[0] = *(const uint4*)qp;
        qf.q[1] = *(const uint4*)(qp + 16);
    }

    float mrow[8], lrow[8];
    v8f acc0, acc1;
#pragma unroll
    for (int v = 0; v < 8; ++v) { mrow[v] = -3.4e38f; lrow[v] = 0.f; acc0[v] = 0.f; acc1[v] = 0.f; }

    const float scale = 0.17677669529663687f;   // 1/sqrt(32)
    int rIdx = (lane >= 16) ? 8 : 0;

    for (int j = 0; j < SEQLEN; j += 32) {
        // stage V^T tile: each lane handles one key row (32 dh values)
        {
            const u16* vp = base + (size_t)(j + lane) * (3 * D_MODEL) + 2 * D_MODEL + h * DHEAD;
#pragma unroll
            for (int dh = 0; dh < 32; ++dh) vt_lds[dh * 32 + lane] = vp[dh];
        }

        // K^T B-fragments for two 16-key score tiles (direct from global:
        // [s][dh] memory layout == NxK fragment layout)
        Frag16 kf0, kf1;
        {
            int kb = (lane >> 4) * 8;
            const u16* kp0 = base + (size_t)(j + (lane & 15)) * (3 * D_MODEL) + D_MODEL + h * DHEAD + kb;
            const u16* kp1 = base + (size_t)(j + 16 + (lane & 15)) * (3 * D_MODEL) + D_MODEL + h * DHEAD + kb;
            kf0.q[0] = *(const uint4*)kp0;  kf0.q[1] = *(const uint4*)(kp0 + 16);
            kf1.q[0] = *(const uint4*)kp1;  kf1.q[1] = *(const uint4*)(kp1 + 16);
        }
        v8f sc0, sc1;
#pragma unroll
        for (int v = 0; v < 8; ++v) { sc0[v] = 0.f; sc1[v] = 0.f; }
        sc0 = wmma_bf16(qf.v, kf0.v, sc0);
        sc1 = wmma_bf16(qf.v, kf1.v, sc1);

        __syncthreads();   // V^T staged before use; P buffer free

        // online softmax, write P (bf16) to LDS
#pragma unroll
        for (int v = 0; v < 8; ++v) {
            float a0 = sc0[v] * scale;
            float a1 = sc1[v] * scale;
            float t  = fmaxf(a0, a1);
#pragma unroll
            for (int off = 8; off > 0; off >>= 1) t = fmaxf(t, __shfl_xor(t, off));
            float mn    = fmaxf(mrow[v], t);
            float alpha = expf(mrow[v] - mn);
            float p0 = expf(a0 - mn);
            float p1 = expf(a1 - mn);
            float ps = p0 + p1;
#pragma unroll
            for (int off = 8; off > 0; off >>= 1) ps += __shfl_xor(ps, off);
            lrow[v] = lrow[v] * alpha + ps;
            mrow[v] = mn;
            acc0[v] *= alpha;
            acc1[v] *= alpha;
            int r = v + rIdx;
            p_lds[r * 32 + (lane & 15)]      = f2bf(p0);
            p_lds[r * 32 + 16 + (lane & 15)] = f2bf(p1);
        }
        __syncthreads();

        // P A-fragment from LDS
        Frag16 pf;
        {
            int m  = lane & 15;
            int kc = (lane >> 4) * 8;
            const u16* pp = p_lds + m * 32 + kc;
            pf.q[0] = *(const uint4*)pp;
            pf.q[1] = *(const uint4*)(pp + 16);
        }
        // V^T B-fragments (dh halves)
        Frag16 vf0, vf1;
        {
            int kb = (lane >> 4) * 8;
            const u16* v0p = vt_lds + (lane & 15) * 32 + kb;
            const u16* v1p = vt_lds + (16 + (lane & 15)) * 32 + kb;
            vf0.q[0] = *(const uint4*)v0p;  vf0.q[1] = *(const uint4*)(v0p + 16);
            vf1.q[0] = *(const uint4*)v1p;  vf1.q[1] = *(const uint4*)(v1p + 16);
        }
        acc0 = wmma_bf16(pf.v, vf0.v, acc0);
        acc1 = wmma_bf16(pf.v, vf1.v, acc1);
        __syncthreads();   // protect LDS reuse next iteration
    }

    // write O (bf16), normalized by row sums
    u16* ob = o + ((size_t)b * SEQLEN + s0) * D_MODEL + h * DHEAD;
#pragma unroll
    for (int v = 0; v < 8; ++v) {
        float inv = 1.f / lrow[v];
        int r = v + rIdx;
        ob[(size_t)r * D_MODEL + (lane & 15)]      = f2bf(acc0[v] * inv);
        ob[(size_t)r * D_MODEL + 16 + (lane & 15)] = f2bf(acc1[v] * inv);
    }
}

// ---------------------------------------------------------------------------
// Host-side orchestration
// ---------------------------------------------------------------------------
extern "C" void kernel_launch(void* const* d_in, const int* in_sizes, int n_in,
                              void* d_out, int out_size, void* d_ws, size_t ws_size,
                              hipStream_t stream) {
    (void)in_sizes; (void)n_in; (void)out_size; (void)ws_size;

    const float* ze      = (const float*)d_in[0];
    const float* cb      = (const float*)d_in[1];
    const float* pos     = (const float*)d_in[2];
    const float* Wqkv    = (const float*)d_in[3];
    const float* bqkv    = (const float*)d_in[4];
    const float* Wo      = (const float*)d_in[5];
    const float* bo      = (const float*)d_in[6];
    const float* ln1_s   = (const float*)d_in[7];
    const float* ln1_b   = (const float*)d_in[8];
    const float* W1      = (const float*)d_in[9];
    const float* b1      = (const float*)d_in[10];
    const float* W2      = (const float*)d_in[11];
    const float* b2      = (const float*)d_in[12];
    const float* ln2_s   = (const float*)d_in[13];
    const float* ln2_b   = (const float*)d_in[14];
    const float* Wout    = (const float*)d_in[15];
    const float* bout    = (const float*)d_in[16];
    float* out           = (float*)d_out;

    // workspace layout
    char* ws = (char*)d_ws;
    size_t off = 0;
    auto take = [&](size_t bytes) { char* p = ws + off; off += (bytes + 255) & ~(size_t)255; return p; };

    u16*   zeb   = (u16*)  take((size_t)MTOK * D_MODEL * 2);
    u16*   cbb   = (u16*)  take((size_t)K_CODES * D_MODEL * 2);
    float* cnorm = (float*)take((size_t)K_CODES * 4);
    int*   idx   = (int*)  take((size_t)MTOK * 4);
    float* x     = (float*)take((size_t)MTOK * D_MODEL * 4);
    u16*   hb    = (u16*)  take((size_t)MTOK * D_MODEL * 2);
    u16*   qkvb  = (u16*)  take((size_t)MTOK * 3 * D_MODEL * 2);
    u16*   ob    = (u16*)  take((size_t)MTOK * D_MODEL * 2);
    u16*   ffb   = (u16*)  take((size_t)MTOK * DFF * 2);
    u16*   xb    = (u16*)  take((size_t)MTOK * D_MODEL * 2);
    u16*   wqkvb = (u16*)  take((size_t)N_BLOCKS * D_MODEL * 3 * D_MODEL * 2);
    u16*   wob   = (u16*)  take((size_t)N_BLOCKS * D_MODEL * D_MODEL * 2);
    u16*   w1b   = (u16*)  take((size_t)N_BLOCKS * D_MODEL * DFF * 2);
    u16*   w2b   = (u16*)  take((size_t)N_BLOCKS * DFF * D_MODEL * 2);
    u16*   woutb = (u16*)  take((size_t)D_MODEL * D_PATCH * 2);

    auto cvt = [&](const float* src, u16* dst, int n) {
        cvt_f32_bf16_kernel<<<n / 256, 256, 0, stream>>>(src, dst);
    };

    // --- precision casts ---
    cvt(ze,   zeb,   MTOK * D_MODEL);
    cvt(cb,   cbb,   K_CODES * D_MODEL);
    cvt(Wqkv, wqkvb, N_BLOCKS * D_MODEL * 3 * D_MODEL);
    cvt(Wo,   wob,   N_BLOCKS * D_MODEL * D_MODEL);
    cvt(W1,   w1b,   N_BLOCKS * D_MODEL * DFF);
    cvt(W2,   w2b,   N_BLOCKS * DFF * D_MODEL);
    cvt(Wout, woutb, D_MODEL * D_PATCH);

    // --- VQ ---
    cb_norms_kernel<<<K_CODES / 8, 256, 0, stream>>>(cb, cnorm);
    vq_argmin_kernel<<<MTOK / 16, 128, 0, stream>>>(zeb, cbb, cnorm, idx);
    gather_pos_kernel<<<(MTOK * D_MODEL) / 256, 256, 0, stream>>>(idx, cb, pos, x);

    // --- transformer blocks ---
    for (int i = 0; i < N_BLOCKS; ++i) {
        layernorm_kernel<<<MTOK / 8, 256, 0, stream>>>(
            x, ln1_s + i * D_MODEL, ln1_b + i * D_MODEL, hb);

        gemm_bf16_kernel<false, false, true, false>
            <<<dim3((3 * D_MODEL) / 64, MTOK / 64), 128, 0, stream>>>(
                hb, wqkvb + (size_t)i * D_MODEL * 3 * D_MODEL,
                bqkv + (size_t)i * 3 * D_MODEL,
                nullptr, nullptr, qkvb, MTOK, 3 * D_MODEL, D_MODEL);

        attention_kernel<<<dim3(SEQLEN / 16, N_HEADS, BATCH), 32, 0, stream>>>(qkvb, ob);

        gemm_bf16_kernel<true, false, false, true>
            <<<dim3(D_MODEL / 64, MTOK / 64), 128, 0, stream>>>(
                ob, wob + (size_t)i * D_MODEL * D_MODEL,
                bo + (size_t)i * D_MODEL,
                x, x, nullptr, MTOK, D_MODEL, D_MODEL);

        layernorm_kernel<<<MTOK / 8, 256, 0, stream>>>(
            x, ln2_s + i * D_MODEL, ln2_b + i * D_MODEL, hb);

        gemm_bf16_kernel<false, true, true, false>
            <<<dim3(DFF / 64, MTOK / 64), 128, 0, stream>>>(
                hb, w1b + (size_t)i * D_MODEL * DFF,
                b1 + (size_t)i * DFF,
                nullptr, nullptr, ffb, MTOK, DFF, D_MODEL);

        gemm_bf16_kernel<true, false, false, true>
            <<<dim3(D_MODEL / 64, MTOK / 64), 128, 0, stream>>>(
                ffb, w2b + (size_t)i * DFF * D_MODEL,
                b2 + (size_t)i * D_MODEL,
                x, x, nullptr, MTOK, D_MODEL, DFF);
    }

    // --- output projection ---
    cvt(x, xb, MTOK * D_MODEL);
    gemm_bf16_kernel<false, false, false, true>
        <<<dim3(D_PATCH / 64, MTOK / 64), 128, 0, stream>>>(
            xb, woutb, bout, nullptr, out, nullptr, MTOK, D_PATCH, D_MODEL);
}
